// VectorQuantizer_76063870812726
// MI455X (gfx1250) — matrix operand order repaired and checked by reading, not compiled, not run on hardware
//
#include <hip/hip_runtime.h>
#include <hip/hip_bf16.h>

typedef __attribute__((ext_vector_type(16))) __bf16 v16bf;
typedef __attribute__((ext_vector_type(4)))  __bf16 v4bf;
typedef __attribute__((ext_vector_type(8)))  float  v8f;
typedef __attribute__((ext_vector_type(4)))  unsigned int u32x4;
typedef __attribute__((ext_vector_type(4)))  int i32x4;
typedef __attribute__((ext_vector_type(8)))  int i32x8;

#define DDIM   256
#define KCODES 1024
#define NROWS  65536
#define TILES  (KCODES / 16)       // 64 code tiles of 16
#define ROWS_PER_WAVE 32           // two 16-row WMMA tiles per wave
#define ROWS_PER_BLOCK 256         // 8 waves * 32 rows
#define NBLOCKS (NROWS / ROWS_PER_BLOCK)  // 256

#if defined(__HIP_DEVICE_COMPILE__) && __has_builtin(__builtin_amdgcn_tensor_load_to_lds)
#define HAVE_TDM 1
#else
#define HAVE_TDM 0
#endif

#if HAVE_TDM
// Issue one TDM copy: 8KB (16 codes x 256 bf16) from global -> LDS. 1-D tile.
__device__ __forceinline__ void tdm_load_tile(unsigned int lds_off, const __bf16* gptr) {
    unsigned long long ga = (unsigned long long)(size_t)(const void*)gptr;
    u32x4 g0;
    g0[0] = 1u;                                        // count=1, user descriptor
    g0[1] = lds_off;                                   // lds_addr (bytes)
    g0[2] = (unsigned int)(ga & 0xFFFFFFFFu);          // global_addr[31:0]
    g0[3] = (unsigned int)((ga >> 32) & 0x1FFFFFFu)    // global_addr[56:32]
          | (2u << 30);                                // type = 2 ("image")
    i32x8 g1;
    g1[0] = (1 << 16);                 // workgroup_mask=0, data_size=1 (2 bytes)
    g1[1] = (int)((262144u & 0xFFFFu) << 16);  // tensor_dim0[15:0] (=0x40000 -> lo16=0)
    g1[2] = (int)(262144u >> 16);              // tensor_dim0[31:16]=4 ; tensor_dim1 lo=0
    g1[3] = (int)(4096u << 16);                // tile_dim0 = 4096 elements
    g1[4] = 0;                                 // tile_dim1=0 (1-D), tile_dim2=0
    g1[5] = (int)262144;                       // tensor_dim0_stride lo32
    g1[6] = 0;
    g1[7] = 0;
    i32x4 gz = {0, 0, 0, 0};
#if __clang_major__ >= 23
    i32x8 gz8 = {0, 0, 0, 0, 0, 0, 0, 0};
    __builtin_amdgcn_tensor_load_to_lds(g0, g1, gz, gz, gz8, 0);
#else
    __builtin_amdgcn_tensor_load_to_lds(g0, g1, gz, gz, 0);
#endif
}
#endif

// ---- prep: convert codebook f32 -> bf16 once, and |e_k|^2 table. grid 64 x 256.
__global__ __launch_bounds__(256)
void vq_prep(const float* __restrict__ emb, __bf16* __restrict__ embbf,
             float* __restrict__ s_sq) {
    const int gtid = blockIdx.x * 256 + threadIdx.x;   // 16384 threads
    const int code = gtid >> 4;
    const int seg  = gtid & 15;                        // 16 f32 each
    const float4* src = reinterpret_cast<const float4*>(emb + (size_t)code * DDIM + seg * 16);
    v4bf* dst = reinterpret_cast<v4bf*>(embbf + (size_t)code * DDIM + seg * 16);
    float lsq = 0.f;
    #pragma unroll
    for (int q = 0; q < 4; ++q) {
        float4 v = src[q];
        lsq += v.x * v.x + v.y * v.y + v.z * v.z + v.w * v.w;
        v4bf w;
        w[0] = (__bf16)v.x; w[1] = (__bf16)v.y;
        w[2] = (__bf16)v.z; w[3] = (__bf16)v.w;
        dst[q] = w;
    }
    #pragma unroll
    for (int m = 8; m >= 1; m >>= 1) lsq += __shfl_xor(lsq, m, 32);
    if (seg == 0) s_sq[code] = lsq;
}

template <bool PRE>
__global__ __launch_bounds__(256)
void vq_argmin_kernel(const float* __restrict__ z,
                      const float* __restrict__ emb,
                      const __bf16* __restrict__ embbf,   // used when PRE
                      const float* __restrict__ s_sq_g,   // used when PRE
                      float* __restrict__ out_q,
                      float* __restrict__ out_idx,
                      float* __restrict__ ws_partial)
{
    __shared__ __align__(32) __bf16 ebuf[2][16 * DDIM];   // 2 x 8KB
    __shared__ float s_tile[2][16];                       // !PRE: per-tile |e|^2
    __shared__ float s_all[KCODES];                       // PRE: full |e|^2 table

    const int tid    = threadIdx.x;
    const int wave   = tid >> 5;
    const int lane   = tid & 31;
    const int lane16 = lane & 15;
    const bool hi    = lane >= 16;
    const int rowBase = blockIdx.x * ROWS_PER_BLOCK + wave * ROWS_PER_WAVE;

    // ---- !PRE staging: convert code tile t (16 x 256 f32 -> bf16) into LDS + |e|^2
    auto stage_cvt = [&](int t, int bufsel) {
        const int crow = tid >> 4;
        const int seg  = tid & 15;
        const float* src = emb + (size_t)(t * 16 + crow) * DDIM + seg * 16;
        __bf16* dst = &ebuf[bufsel][crow * DDIM + seg * 16];
        float lsq = 0.f;
        const float4* s4 = reinterpret_cast<const float4*>(src);
        #pragma unroll
        for (int q = 0; q < 4; ++q) {
            float4 v = s4[q];
            lsq += v.x * v.x + v.y * v.y + v.z * v.z + v.w * v.w;
            v4bf w;
            w[0] = (__bf16)v.x; w[1] = (__bf16)v.y;
            w[2] = (__bf16)v.z; w[3] = (__bf16)v.w;
            *reinterpret_cast<v4bf*>(dst + q * 4) = w;
        }
        #pragma unroll
        for (int m = 8; m >= 1; m >>= 1) lsq += __shfl_xor(lsq, m, 32);
        if (seg == 0) s_tile[bufsel][crow] = lsq;
    };

    // ---- PRE staging: raw 8KB bf16 tile copy (TDM if available, else b128 copy)
    auto stage_pre = [&](int t, int bufsel) {
#if HAVE_TDM
        if (wave == 0) {
            unsigned int lds_off = (unsigned int)(size_t)(void*)&ebuf[bufsel][0];
            tdm_load_tile(lds_off, embbf + (size_t)t * 16 * DDIM);
        }
#else
        const uint4* s = reinterpret_cast<const uint4*>(
            reinterpret_cast<const char*>(embbf + (size_t)t * 16 * DDIM) + tid * 32);
        uint4* d = reinterpret_cast<uint4*>(
            reinterpret_cast<char*>(&ebuf[bufsel][0]) + tid * 32);
        d[0] = s[0]; d[1] = s[1];
#endif
    };

    if (PRE) {   // pull |e_k|^2 table into LDS once
        #pragma unroll
        for (int j = 0; j < 4; ++j) s_all[tid + 256 * j] = s_sq_g[tid + 256 * j];
    }

    // ---- load A operands for two 16-row tiles: value = -2*z, bf16 (contiguous k order, same as B)
    // lane l<16: M=l, K {0..15}; lane l>=16: M=l-16, K {16..31}
    v16bf A0[8], A1[8];
    {
        const int ofsA = hi ? 16 : 0;
        #pragma unroll
        for (int half = 0; half < 2; ++half) {
            const int myRow = rowBase + half * 16 + lane16;
            const float* zr = z + (size_t)myRow * DDIM;
            #pragma unroll
            for (int c = 0; c < 8; ++c) {
                const float* p = zr + c * 32 + ofsA;
                float4 u0 = *reinterpret_cast<const float4*>(p);
                float4 u1 = *reinterpret_cast<const float4*>(p + 4);
                float4 u2 = *reinterpret_cast<const float4*>(p + 8);
                float4 u3 = *reinterpret_cast<const float4*>(p + 12);
                v16bf a;
                a[0]  = (__bf16)(-2.f * u0.x); a[1]  = (__bf16)(-2.f * u0.y);
                a[2]  = (__bf16)(-2.f * u0.z); a[3]  = (__bf16)(-2.f * u0.w);
                a[4]  = (__bf16)(-2.f * u1.x); a[5]  = (__bf16)(-2.f * u1.y);
                a[6]  = (__bf16)(-2.f * u1.z); a[7]  = (__bf16)(-2.f * u1.w);
                a[8]  = (__bf16)(-2.f * u2.x); a[9]  = (__bf16)(-2.f * u2.y);
                a[10] = (__bf16)(-2.f * u2.z); a[11] = (__bf16)(-2.f * u2.w);
                a[12] = (__bf16)(-2.f * u3.x); a[13] = (__bf16)(-2.f * u3.y);
                a[14] = (__bf16)(-2.f * u3.z); a[15] = (__bf16)(-2.f * u3.w);
                if (half == 0) A0[c] = a; else A1[c] = a;
            }
        }
    }

    float best0[8], best1[8];
    int   bidx0[8], bidx1[8];
    #pragma unroll
    for (int r = 0; r < 8; ++r) {
        best0[r] = 3.4e38f; bidx0[r] = 0;
        best1[r] = 3.4e38f; bidx1[r] = 0;
    }

    if (PRE) stage_pre(0, 0); else stage_cvt(0, 0);
#if HAVE_TDM
    if (PRE) __builtin_amdgcn_s_wait_tensorcnt(0);   // no-op for non-issuing waves
#endif
    __syncthreads();

    const int ofsB = hi ? 16 : 0;   // B layout: lane<16 -> K0..15, lane>=16 -> K16..31
    for (int t = 0; t < TILES; ++t) {
        if (t + 1 < TILES) {
            if (PRE) stage_pre(t + 1, (t + 1) & 1);
            else     stage_cvt(t + 1, (t + 1) & 1);
        }

        const __bf16* eb = &ebuf[t & 1][0] + lane16 * DDIM + ofsB;
        const float sq = PRE ? s_all[t * 16 + lane16] : s_tile[t & 1][lane16];
        v8f acc0, acc1;
        #pragma unroll
        for (int r = 0; r < 8; ++r) { acc0[r] = sq; acc1[r] = sq; }  // C init = |e_k|^2

        // software-pipelined B: each B chunk feeds two back-to-back WMMAs
        v16bf bcur = *reinterpret_cast<const v16bf*>(eb);
        #pragma unroll
        for (int c = 0; c < 8; ++c) {
            v16bf bnext = bcur;
            if (c + 1 < 8)
                bnext = *reinterpret_cast<const v16bf*>(eb + (c + 1) * 32);
            acc0 = __builtin_amdgcn_wmma_f32_16x16x32_bf16(
                       false, A0[c], false, bcur, (short)0, acc0, false, false);
            acc1 = __builtin_amdgcn_wmma_f32_16x16x32_bf16(
                       false, A1[c], false, bcur, (short)0, acc1, false, false);
            bcur = bnext;
        }
        const int k = t * 16 + lane16;   // score = |e_k|^2 - 2 z.e_k
        #pragma unroll
        for (int r = 0; r < 8; ++r) {
            if (acc0[r] < best0[r]) { best0[r] = acc0[r]; bidx0[r] = k; }
            if (acc1[r] < best1[r]) { best1[r] = acc1[r]; bidx1[r] = k; }
        }
#if HAVE_TDM
        if (PRE) __builtin_amdgcn_s_wait_tensorcnt(0);
#endif
        __syncthreads();
    }

    // ---- cross-lane argmin within each 16-lane N group (ties -> lower index)
    #pragma unroll
    for (int r = 0; r < 8; ++r) {
        float s0 = best0[r]; int i0 = bidx0[r];
        float s1 = best1[r]; int i1 = bidx1[r];
        #pragma unroll
        for (int m = 8; m >= 1; m >>= 1) {
            float so0 = __shfl_xor(s0, m, 32);
            int   io0 = __shfl_xor(i0, m, 32);
            float so1 = __shfl_xor(s1, m, 32);
            int   io1 = __shfl_xor(i1, m, 32);
            if (so0 < s0 || (so0 == s0 && io0 < i0)) { s0 = so0; i0 = io0; }
            if (so1 < s1 || (so1 == s1 && io1 < i1)) { s1 = so1; i1 = io1; }
        }
        bidx0[r] = i0; bidx1[r] = i1;
    }
    // lanes 0..15 hold winners for rows +r / +16+r ; lanes 16..31 for rows +8+r / +24+r
    if (lane == 0) {
        #pragma unroll
        for (int r = 0; r < 8; ++r) {
            out_idx[rowBase + r]      = (float)bidx0[r];
            out_idx[rowBase + 16 + r] = (float)bidx1[r];
        }
    }
    if (lane == 16) {
        #pragma unroll
        for (int r = 0; r < 8; ++r) {
            out_idx[rowBase + 8 + r]  = (float)bidx0[r];
            out_idx[rowBase + 24 + r] = (float)bidx1[r];
        }
    }
    int rowIdx[32];
    #pragma unroll
    for (int r = 0; r < 8; ++r) {
        rowIdx[r]      = __shfl(bidx0[r], 0, 32);
        rowIdx[8 + r]  = __shfl(bidx0[r], 16, 32);
        rowIdx[16 + r] = __shfl(bidx1[r], 0, 32);
        rowIdx[24 + r] = __shfl(bidx1[r], 16, 32);
    }

    // ---- epilogue: quantized_st = z + (e - z); accumulate (e - z)^2 for loss
    float lsum = 0.f;
    #pragma unroll
    for (int r = 0; r < 32; ++r) {
        const int k   = rowIdx[r];
        const int row = rowBase + r;
        const float4* ep = reinterpret_cast<const float4*>(emb + (size_t)k * DDIM) + lane * 2;
        const float4* zp = reinterpret_cast<const float4*>(z + (size_t)row * DDIM) + lane * 2;
        float4* qp = reinterpret_cast<float4*>(out_q + (size_t)row * DDIM) + lane * 2;
        #pragma unroll
        for (int h = 0; h < 2; ++h) {
            float4 e = ep[h], zz = zp[h], d, q;
            d.x = e.x - zz.x; d.y = e.y - zz.y; d.z = e.z - zz.z; d.w = e.w - zz.w;
            q.x = zz.x + d.x; q.y = zz.y + d.y; q.z = zz.z + d.z; q.w = zz.w + d.w;
            qp[h] = q;
            lsum += d.x * d.x + d.y * d.y + d.z * d.z + d.w * d.w;
        }
    }
    #pragma unroll
    for (int m = 16; m >= 1; m >>= 1) lsum += __shfl_xor(lsum, m, 32);
    if (lane == 0) ws_partial[blockIdx.x * 8 + wave] = lsum;
}

// deterministic tree reduction of 2048 per-wave partials -> vq_loss
__global__ __launch_bounds__(256)
void vq_loss_reduce(const float* __restrict__ ws_partial,
                    float* __restrict__ out_loss)
{
    __shared__ float red[256];
    const int tid = threadIdx.x;
    float s = 0.f;
    #pragma unroll
    for (int j = 0; j < 8; ++j) s += ws_partial[tid + 256 * j];
    red[tid] = s;
    __syncthreads();
    for (int ofs = 128; ofs > 0; ofs >>= 1) {
        if (tid < ofs) red[tid] += red[tid + ofs];
        __syncthreads();
    }
    if (tid == 0) *out_loss = red[0] * (1.25f / 16777216.f);  // (1+0.25)*mean
}

extern "C" void kernel_launch(void* const* d_in, const int* in_sizes, int n_in,
                              void* d_out, int out_size, void* d_ws, size_t ws_size,
                              hipStream_t stream) {
    const float* z   = (const float*)d_in[0];   // 16*256*64*64 f32
    const float* emb = (const float*)d_in[1];   // 1024*256 f32

    float* out      = (float*)d_out;
    float* out_q    = out;                      // quantized_st: 16777216 f32
    float* out_loss = out + (size_t)16777216;   // vq_loss: 1 f32
    float* out_idx  = out_loss + 1;             // indices: 65536 (as float)

    const size_t EMBBF_BYTES = (size_t)KCODES * DDIM * 2;   // 524288
    const size_t SQ_BYTES    = (size_t)KCODES * 4;          // 4096
    const size_t PART_BYTES  = (size_t)NBLOCKS * 8 * 4;     // 8192

    if (ws_size >= EMBBF_BYTES + SQ_BYTES + PART_BYTES) {
        __bf16* embbf = (__bf16*)d_ws;
        float*  s_sq  = (float*)((char*)d_ws + EMBBF_BYTES);
        float*  part  = (float*)((char*)d_ws + EMBBF_BYTES + SQ_BYTES);
        vq_prep<<<KCODES * 16 / 256, 256, 0, stream>>>(emb, embbf, s_sq);
        vq_argmin_kernel<true><<<NBLOCKS, 256, 0, stream>>>(
            z, emb, embbf, s_sq, out_q, out_idx, part);
        vq_loss_reduce<<<1, 256, 0, stream>>>(part, out_loss);
    } else {
        float* part = (float*)d_ws;
        vq_argmin_kernel<false><<<NBLOCKS, 256, 0, stream>>>(
            z, emb, (const __bf16*)nullptr, (const float*)nullptr, out_q, out_idx, part);
        vq_loss_reduce<<<1, 256, 0, stream>>>(part, out_loss);
    }
}
